// Retention_60309930770639
// MI455X (gfx1250) — matrix operand order repaired and checked by reading, not compiled, not run on hardware
//
#include <hip/hip_runtime.h>

// Retention (RetNet) fused kernels for MI455X (gfx1250, wave32, WMMA).
// B=2, S=2048, E=1024, H=16, D=64.

typedef __attribute__((ext_vector_type(16))) __bf16 v16bf;
typedef __attribute__((ext_vector_type(8)))  __bf16 v8bf;
typedef __attribute__((ext_vector_type(8)))  float  v8f;

#define B_ 2
#define S_ 2048
#define E_ 1024
#define H_ 16
#define D_ 64
#define EPS_ 1e-6f

__device__ __forceinline__ v16bf cat8(v8bf lo, v8bf hi) {
    return __builtin_shufflevector(lo, hi, 0,1,2,3,4,5,6,7,8,9,10,11,12,13,14,15);
}

// -----------------------------------------------------------------------------
// Kernel 1: fused QKVG projection.
// 64(M) x 64(N) tile of X(4096x1024) @ W(1024x1024) with bf16 WMMA; epilogue:
//   which==0: q -> theta-shift -> qr[b][h][s][d]           (bf16)
//   which==1: k -> *E^-0.5 -> theta-shift -> kr[b][h][s][d] (bf16)
//   which==2: v -> vT[b][h][d][s]  (transposed for B-fragment loads, bf16)
//   which==3: g -> g[b*S+s][e]     (f32)
// 4 waves / block; wave w owns M-rows [w*16, w*16+16) x all 64 N.
// -----------------------------------------------------------------------------
__global__ __launch_bounds__(128)
void proj_kernel(const float* __restrict__ x,
                 const float* __restrict__ Wq, const float* __restrict__ Wk,
                 const float* __restrict__ Wv, const float* __restrict__ Wg,
                 __bf16* __restrict__ qr, __bf16* __restrict__ kr,
                 __bf16* __restrict__ vT, float* __restrict__ g)
{
    __shared__ __bf16 Alds[64][48];   // [m][k], padded row stride 96B
    __shared__ __bf16 WTlds[64][48];  // [n][k] (transposed W tile)

    const int which = blockIdx.z;
    const float* W = (which == 0) ? Wq : (which == 1) ? Wk : (which == 2) ? Wv : Wg;

    const int mt = blockIdx.y, nt = blockIdx.x;
    const int mbase = mt * 64, nbase = nt * 64;
    const int tid  = threadIdx.x;
    const int wv   = tid >> 5;
    const int lane = tid & 31;
    const int r    = lane & 15;
    const int hi   = lane >> 4;
    const int hi8  = hi * 8;
    const int ks16 = hi * 16;

    v8f cacc[4] = {};

    for (int kk = 0; kk < E_; kk += 32) {
        // Prefetch next K-tile (global_prefetch_b8 on gfx1250).
        const int nk = kk + 32;
        if (nk < E_) {
            if (tid < 64)       __builtin_prefetch(&x[(mbase + tid) * E_ + nk], 0, 1);
            else if (tid < 96)  __builtin_prefetch(&W[(nk + tid - 64) * E_ + nbase], 0, 1);
            else                __builtin_prefetch(&W[(nk + tid - 96) * E_ + nbase + 32], 0, 1);
        }
        // Stage A tile 64x32 (f32 -> bf16), coalesced.
        for (int i = tid; i < 2048; i += 128) {
            int row = i >> 5, kc = i & 31;
            Alds[row][kc] = (__bf16)x[(mbase + row) * E_ + kk + kc];
        }
        // Stage W tile 32x64 transposed -> WT[n][k].
        for (int i = tid; i < 2048; i += 128) {
            int krow = i >> 6, nc = i & 63;
            WTlds[nc][krow] = (__bf16)W[(kk + krow) * E_ + nbase + nc];
        }
        __syncthreads();

        const int arow = wv * 16 + r;
        v16bf a = cat8(*(const v8bf*)&Alds[arow][ks16],
                       *(const v8bf*)&Alds[arow][ks16 + 8]);
        #pragma unroll
        for (int t = 0; t < 4; ++t) {
            v16bf b = *(const v16bf*)&WTlds[t * 16 + r][ks16];
            cacc[t] = __builtin_amdgcn_wmma_f32_16x16x32_bf16(
                false, a, false, b, (short)0, cacc[t], false, false);
        }
        __syncthreads();
    }

    // Epilogue. C layout: VGPR j, lane L -> M = j + 8*hi, N = r.
    const float kscale = (which == 1) ? 0.03125f : 1.0f;  // E^-0.5
    #pragma unroll
    for (int t = 0; t < 4; ++t) {
        const int colg = nbase + t * 16 + r;              // in [0, E)
        const int d  = colg & 63;
        const int hh = colg >> 6;
        // angle[d] = 10000^{-(d/2)/31}; per-lane constant within this t.
        const float ai = __expf(-(float)(d >> 1) * (9.210340371976184f / 31.0f));
        #pragma unroll
        for (int j = 0; j < 8; ++j) {
            float val = cacc[t][j];
            const int rowg = mbase + wv * 16 + j + hi8;   // in [0, B*S)
            if (which == 3) {
                g[rowg * E_ + colg] = val;
            } else if (which == 2) {
                const int bb = rowg >> 11, s = rowg & (S_ - 1);
                vT[((bb * H_ + hh) * D_ + d) * S_ + s] = (__bf16)val;
            } else {
                // theta shift: partner element e^1 lives in lane^1, same VGPR j.
                float partner = __shfl_xor(val, 1, 32);
                const int s  = rowg & (S_ - 1);
                const int bb = rowg >> 11;
                float ph = (float)s * ai;
                float cs = __cosf(ph), sn = __sinf(ph);
                float rot = (d & 1) ? partner : -partner;  // rotate_every_two
                float res = (val * cs + rot * sn) * kscale;
                __bf16* dst = (which == 0) ? qr : kr;
                dst[((bb * H_ + hh) * S_ + s) * D_ + d] = (__bf16)res;
            }
        }
    }
}

// -----------------------------------------------------------------------------
// Kernel 2: streamed retention attention + denom + RMS-norm + silu(g) gate.
// One wave per (b, h, 16-row q tile); streams 32-wide key blocks (causal).
// Decay mask factored: exp((q-k)dec) = T(block) * rowf(j) * colf(tt,lane),
// so the inner loop has zero transcendentals (T updated multiplicatively).
// -----------------------------------------------------------------------------
__global__ __launch_bounds__(128)
void attn_kernel(const __bf16* __restrict__ qr, const __bf16* __restrict__ kr,
                 const __bf16* __restrict__ vT, const float* __restrict__ g,
                 float* __restrict__ out)
{
    __shared__ __bf16 Plds[4][16][48];  // per-wave P bounce, padded stride 96B

    const int tid  = threadIdx.x;
    const int wv   = tid >> 5;
    const int lane = tid & 31;
    const int r    = lane & 15;
    const int hi   = lane >> 4;
    const int hi8  = hi * 8;
    const int ks16 = hi * 16;

    const int wg = blockIdx.x * 4 + wv;        // 4096 waves total
    const int qt = wg & 127;                   // S/16
    const int h  = (wg >> 7) & (H_ - 1);
    const int b  = wg >> 11;
    const int qb = qt * 16;

    const float dec = logf(1.0f - exp2f((float)(-5 - h)));

    // rowf[j] = exp((j+hi8)*dec) / sqrt(rowsum(q))   (geometric-series rowsum)
    float rowf[8];
    #pragma unroll
    for (int j = 0; j < 8; ++j) {
        int q = qb + j + hi8;
        float rs = (1.0f - __expf((float)(q + 1) * dec)) / (1.0f - __expf(dec));
        rowf[j] = __expf((float)(j + hi8) * dec) * rsqrtf(rs);
    }
    const float colf0 = __expf(-(float)r * dec);         // exponents <= ~1
    const float colf1 = __expf(-(float)(16 + r) * dec);
    float       T     = __expf((float)qb * dec);         // >= e^-65, no underflow
    const float Tup   = __expf(-32.0f * dec);

    const __bf16* qp = qr + (size_t)(b * H_ + h) * S_ * D_;
    const __bf16* kp = kr + (size_t)(b * H_ + h) * S_ * D_;
    const __bf16* vp = vT + (size_t)(b * H_ + h) * D_ * S_;

    // q A-fragments (held for entire key loop).
    const __bf16* qrow = qp + (qb + r) * D_;
    v16bf aq0 = cat8(*(const v8bf*)(qrow + ks16),      *(const v8bf*)(qrow + ks16 + 8));
    v16bf aq1 = cat8(*(const v8bf*)(qrow + 32 + ks16), *(const v8bf*)(qrow + 32 + ks16 + 8));

    v8f   acc[4]    = {};
    float rowabs[8] = {};

    const int nkb = ((qb + 15) >> 5) + 1;
    for (int kb = 0; kb < nkb; ++kb) {
        const int kbase = kb * 32;
        // Prefetch next key block (kr rows + vT rows).
        if (kbase + 32 <= qb) {
            __builtin_prefetch(kp + (kbase + 32 + lane) * D_, 0, 1);
            __builtin_prefetch(vp + lane * S_ + kbase + 32, 0, 1);
            __builtin_prefetch(vp + (lane + 32) * S_ + kbase + 32, 0, 1);
        }
        #pragma unroll
        for (int tt = 0; tt < 2; ++tt) {
            const int kt = kbase + tt * 16;
            if (kt <= qb + 15) {                 // tile has at least one valid kg
                const __bf16* krow = kp + (kt + r) * D_;
                v16bf bk0 = *(const v16bf*)(krow + ks16);        // d 0..31
                v16bf bk1 = *(const v16bf*)(krow + 32 + ks16);   // d 32..63
                v8f sc = {};
                sc = __builtin_amdgcn_wmma_f32_16x16x32_bf16(false, aq0, false, bk0, (short)0, sc, false, false);
                sc = __builtin_amdgcn_wmma_f32_16x16x32_bf16(false, aq1, false, bk1, (short)0, sc, false, false);
                const float tfac = T * (tt ? colf1 : colf0);
                if (kt + 15 <= qb) {
                    // fully below diagonal: no causal select needed
                    #pragma unroll
                    for (int j = 0; j < 8; ++j) {
                        float sv = sc[j] * (rowf[j] * tfac);
                        rowabs[j] += fabsf(sv);
                        Plds[wv][j + hi8][tt * 16 + r] = (__bf16)sv;
                    }
                } else {
                    const int diffbase = qb + hi8 - kt - r;   // qg-kg = diffbase+j
                    #pragma unroll
                    for (int j = 0; j < 8; ++j) {
                        float sv = sc[j] * (rowf[j] * tfac);
                        sv = (diffbase + j >= 0) ? sv : 0.0f;
                        rowabs[j] += fabsf(sv);
                        Plds[wv][j + hi8][tt * 16 + r] = (__bf16)sv;
                    }
                }
            } else {
                // fully above diagonal: zero P (still part of the K=32 WMMA)
                #pragma unroll
                for (int j = 0; j < 8; ++j)
                    Plds[wv][j + hi8][tt * 16 + r] = (__bf16)0.0f;
            }
        }
        // P 16x32 as A-fragment (LDS in-order within a wave).
        v16bf ap = cat8(*(const v8bf*)&Plds[wv][r][ks16],
                        *(const v8bf*)&Plds[wv][r][ks16 + 8]);
        #pragma unroll
        for (int t = 0; t < 4; ++t) {
            v16bf bv = *(const v16bf*)(vp + (t * 16 + r) * S_ + kbase + ks16);
            acc[t] = __builtin_amdgcn_wmma_f32_16x16x32_bf16(
                false, ap, false, bv, (short)0, acc[t], false, false);
        }
        T *= Tup;
    }

    // Reduce |qk| row sums across the 16 lanes holding each row's N values.
    #pragma unroll
    for (int j = 0; j < 8; ++j) {
        float v = rowabs[j];
        v += __shfl_xor(v, 1, 32);
        v += __shfl_xor(v, 2, 32);
        v += __shfl_xor(v, 4, 32);
        v += __shfl_xor(v, 8, 32);
        rowabs[j] = v;
    }
    float dscale[8];
    #pragma unroll
    for (int j = 0; j < 8; ++j)
        dscale[j] = 1.0f / fminf(fmaxf(rowabs[j], 1.0f), 50000.0f);

    // Apply denom, gather sum of squares for RMS norm over D=64.
    float ssq[8] = {};
    #pragma unroll
    for (int t = 0; t < 4; ++t)
        #pragma unroll
        for (int j = 0; j < 8; ++j) {
            float v = acc[t][j] * dscale[j];
            acc[t][j] = v;
            ssq[j] += v * v;
        }
    #pragma unroll
    for (int j = 0; j < 8; ++j) {
        float v = ssq[j];
        v += __shfl_xor(v, 1, 32);
        v += __shfl_xor(v, 2, 32);
        v += __shfl_xor(v, 4, 32);
        v += __shfl_xor(v, 8, 32);
        ssq[j] = rsqrtf(v * (1.0f / (float)D_) + EPS_);
    }

    // silu(g) gate and final store.
    #pragma unroll
    for (int t = 0; t < 4; ++t)
        #pragma unroll
        for (int j = 0; j < 8; ++j) {
            const int qg  = qb + j + hi8;
            const int e   = h * D_ + t * 16 + r;
            const int idx = (b * S_ + qg) * E_ + e;
            float gv = g[idx];
            float si = gv / (1.0f + __expf(-gv));
            out[idx] = si * (acc[t][j] * ssq[j]);
        }
}

// -----------------------------------------------------------------------------
extern "C" void kernel_launch(void* const* d_in, const int* in_sizes, int n_in,
                              void* d_out, int out_size, void* d_ws, size_t ws_size,
                              hipStream_t stream) {
    const float* x  = (const float*)d_in[0];
    const float* Wq = (const float*)d_in[1];
    const float* Wk = (const float*)d_in[2];
    const float* Wv = (const float*)d_in[3];
    const float* Wg = (const float*)d_in[4];

    // Workspace layout: qr(8MB) | kr(8MB) | vT(8MB) | g(16MB)  => 40 MB total.
    const size_t QKV = (size_t)B_ * H_ * S_ * D_;
    __bf16* qr = (__bf16*)d_ws;
    __bf16* kr = qr + QKV;
    __bf16* vT = kr + QKV;
    float*  gg = (float*)(vT + QKV);
    float*  out = (float*)d_out;

    proj_kernel<<<dim3(E_ / 64, (B_ * S_) / 64, 4), 128, 0, stream>>>(
        x, Wq, Wk, Wv, Wg, qr, kr, vT, gg);

    attn_kernel<<<dim3((B_ * H_ * (S_ / 16)) / 4), 128, 0, stream>>>(
        qr, kr, vT, gg, out);
}
